// ToposMambaBlock_67765993996447
// MI455X (gfx1250) — matrix operand order repaired and checked by reading, not compiled, not run on hardware
//
#include <hip/hip_runtime.h>
#include <hip/hip_bf16.h>

typedef __attribute__((ext_vector_type(16))) _Float16 v16h;
typedef __attribute__((ext_vector_type(8)))  float    v8f;

#define B_BATCH 8
#define SEQ     4096
#define D_MODEL 1024
#define D_STATE 64
#define EPS     1e-6f

// LDS byte offsets inside the scan kernel's dynamic LDS block
// (no static __shared__ vars, so dynamic LDS starts at offset 0).
#define H_BYTES   (D_MODEL * D_STATE * 4)   // 262144
#define UBUF0_OFF (H_BYTES)                 // 64 floats
#define UBUF1_OFF (H_BYTES + 256)           // 64 floats
#define RED_OFF   (H_BYTES + 512)           // 33 floats

__device__ __forceinline__ float sigmoidf_(float v) {
    return 1.0f / (1.0f + __expf(-v));
}
__device__ __forceinline__ float clamp01(float v) {
    return fminf(fmaxf(v, 0.0f), 1.0f);
}

// ---------------------------------------------------------------------------
// Phase 0: parameter prep. sigmoid() all weights once.
//   A_s [D,N] f32, C_s [D,N] f32, g_s [D] f32, Bm_h [N,D] f16
// ---------------------------------------------------------------------------
__global__ __launch_bounds__(256) void prep_kernel(
    const float* __restrict__ A_raw, const float* __restrict__ B_w,
    const float* __restrict__ C_w,   const float* __restrict__ gamma,
    float* __restrict__ A_s, _Float16* __restrict__ Bm_h,
    float* __restrict__ C_s, float* __restrict__ g_s)
{
    int i = blockIdx.x * blockDim.x + threadIdx.x;
    if (i < D_MODEL * D_STATE) {
        A_s[i]  = sigmoidf_(A_raw[i]);
        C_s[i]  = sigmoidf_(C_w[i]);
        Bm_h[i] = (_Float16)sigmoidf_(B_w[i]);
    }
    if (i < D_MODEL) g_s[i] = sigmoidf_(gamma[i]);
}

// ---------------------------------------------------------------------------
// Phase 1: U[bs, n] = sum_d x[bs, d] * Bm[n, d]   (M=32768, K=1024, N=64)
// v_wmma_f32_16x16x32_f16. One wave -> 16x64 output tile (4 accumulators).
// 8 waves per block -> 128 rows per block, 256 blocks.
// ---------------------------------------------------------------------------
__global__ __launch_bounds__(256) void gemm_u_kernel(
    const float* __restrict__ x, const _Float16* __restrict__ Bm,
    float* __restrict__ U)
{
    const int K    = D_MODEL;
    const int lane = threadIdx.x & 31;
    const int wave = threadIdx.x >> 5;                 // 0..7
    const long rowBase = (long)blockIdx.x * 128 + wave * 16;
    const int m    = lane & 15;                        // row within 16-row tile
    const int half = lane >> 4;                        // lane-group 0/1

    v8f acc0 = {}, acc1 = {}, acc2 = {}, acc3 = {};

    const float* xrow = x + (rowBase + m) * (long)K;

    for (int kt = 0; kt < K; kt += 32) {
        // ---- A fragment (16x32 f16): contiguous K order matching B —
        // lanes 0-15 (half=0): element s holds K = s
        // lanes 16-31 (half=1): K = 16 + s. Convert f32 x on the fly.
        v16h a;
        const float* xr = xrow + kt + half * 16;
        #pragma unroll
        for (int v = 0; v < 8; ++v) {
            int k = v << 1;
            a[2 * v]     = (_Float16)xr[k];
            a[2 * v + 1] = (_Float16)xr[k + 1];
        }
        // ---- B fragments (32x16 f16 each): lanes 0-15 hold K=0..15,
        // lanes 16-31 hold K=16..31, N = lane%16 -> one contiguous,
        // 32B-aligned v16h load per lane from Bm[n, kt + 16*half ..].
        #pragma unroll
        for (int j = 0; j < 4; ++j) {
            int n = j * 16 + m;
            v16h b = *(const v16h*)(Bm + (long)n * K + kt + half * 16);
            switch (j) {
            case 0: acc0 = __builtin_amdgcn_wmma_f32_16x16x32_f16(
                        false, a, false, b, (short)0, acc0, false, false); break;
            case 1: acc1 = __builtin_amdgcn_wmma_f32_16x16x32_f16(
                        false, a, false, b, (short)0, acc1, false, false); break;
            case 2: acc2 = __builtin_amdgcn_wmma_f32_16x16x32_f16(
                        false, a, false, b, (short)0, acc2, false, false); break;
            case 3: acc3 = __builtin_amdgcn_wmma_f32_16x16x32_f16(
                        false, a, false, b, (short)0, acc3, false, false); break;
            }
        }
    }

    // ---- Store C/D: VGPR r, lanes 0-15 -> M = r, lanes 16-31 -> M = r+8.
    #pragma unroll
    for (int r = 0; r < 8; ++r) {
        long row = rowBase + r + half * 8;
        float* Ur = U + row * D_STATE;
        Ur[0 * 16 + m] = acc0[r];
        Ur[1 * 16 + m] = acc1[r];
        Ur[2 * 16 + m] = acc2[r];
        Ur[3 * 16 + m] = acc3[r];
    }
}

// ---------------------------------------------------------------------------
// Phase 2: sequential scan. One workgroup per batch; h[b] (1024x64 f32 =
// 256 KB) resident in LDS for all 4096 steps (CDNA5: 320 KB/WGP).
// Thread d owns h[d, 0..63]. A/Cm rows streamed from L2 as float4.
// U rows are double-buffered in LDS and prefetched with the CDNA5 async
// path (global_load_async_to_lds_b64 / ASYNCcnt) so the serial per-step
// dependency never waits on global memory.
// ---------------------------------------------------------------------------
__global__ __launch_bounds__(1024) void scan_kernel(
    const float* __restrict__ x,   const float* __restrict__ A_s,
    const float* __restrict__ C_s, const float* __restrict__ g_s,
    const float* __restrict__ U,
    float* __restrict__ y_out, float* __restrict__ h_out)
{
    extern __shared__ float smem[];
    float* h    = smem;                                  // 1024*64 floats
    float* ubuf0 = smem + (UBUF0_OFF / 4);               // 64 floats
    float* ubuf1 = smem + (UBUF1_OFF / 4);               // 64 floats
    float* red  = smem + (RED_OFF / 4);                  // 33 floats

    const int b    = blockIdx.x;
    const int d    = threadIdx.x;
    const int lane = d & 31;
    const int wv   = d >> 5;

    float4* hrow = (float4*)(h + d * D_STATE);
    #pragma unroll
    for (int q = 0; q < 16; ++q) hrow[q] = make_float4(0.f, 0.f, 0.f, 0.f);

    const float4* Arow = (const float4*)(A_s + d * D_STATE);
    const float4* Crow = (const float4*)(C_s + d * D_STATE);
    const float gd = g_s[d];

    const long  xbase = (long)b * SEQ * D_MODEL;
    const float* Ub   = U + (long)b * SEQ * D_STATE;

    // Synchronously preload U row for t = 0 into buffer 0.
    if (d < D_STATE) ubuf0[d] = Ub[d];
    __syncthreads();

    for (int t = 0; t < SEQ; ++t) {
        const float* ucur = (t & 1) ? ubuf1 : ubuf0;

        // Async prefetch of U row t+1 into the other LDS buffer.
        // Wave 0 only: 32 lanes x b64 = the full 256-byte row.
        if (d < 32 && (t + 1) < SEQ) {
            const unsigned lds_off =
                (((t + 1) & 1) ? UBUF1_OFF : UBUF0_OFF) + (unsigned)lane * 8u;
            const float* gp = Ub + (size_t)(t + 1) * D_STATE + lane * 2;
            asm volatile("global_load_async_to_lds_b64 %0, %1, off"
                         :: "v"(lds_off),
                            "v"((unsigned long long)(uintptr_t)gp)
                         : "memory");
        }

        const float xd = x[xbase + (long)t * D_MODEL + d];
        if (t + 1 < SEQ)
            __builtin_prefetch(&x[xbase + (long)(t + 1) * D_MODEL + d], 0, 0);

        float yd = 0.0f;
        #pragma unroll
        for (int q = 0; q < 16; ++q) {
            float4 hv = hrow[q];
            float4 a  = Arow[q];
            float4 c  = Crow[q];
            float u0 = ucur[4 * q + 0], u1 = ucur[4 * q + 1];
            float u2 = ucur[4 * q + 2], u3 = ucur[4 * q + 3];
            hv.x = clamp01(fmaf(hv.x, a.x, xd * u0));
            hv.y = clamp01(fmaf(hv.y, a.y, xd * u1));
            hv.z = clamp01(fmaf(hv.z, a.z, xd * u2));
            hv.w = clamp01(fmaf(hv.w, a.w, xd * u3));
            hrow[q] = hv;
            yd = fmaf(hv.x, c.x, yd);
            yd = fmaf(hv.y, c.y, yd);
            yd = fmaf(hv.z, c.z, yd);
            yd = fmaf(hv.w, c.w, yd);
        }

        // L1 norm over D: wave32 shuffle reduce, then 32-wave LDS tree.
        float s = yd;
        #pragma unroll
        for (int off = 16; off > 0; off >>= 1) s += __shfl_down(s, off, 32);
        if (lane == 0) red[wv] = s;
        __syncthreads();
        if (d < 32) {
            float r = red[d];
            #pragma unroll
            for (int off = 16; off > 0; off >>= 1) r += __shfl_down(r, off, 32);
            if (d == 0) red[32] = r;
            // Wave 0 issued the async prefetch; retire it before the
            // barrier so every wave may read the buffer next iteration.
#if __has_builtin(__builtin_amdgcn_s_wait_asynccnt)
            __builtin_amdgcn_s_wait_asynccnt(0);
#else
            asm volatile("s_wait_asynccnt 0x0" ::: "memory");
#endif
        }
        __syncthreads();
        const float tot = red[32];

        float yn = gd * yd / (tot + EPS);
        y_out[xbase + (long)t * D_MODEL + d] = clamp01(yn + xd);
    }

    // h_final [B, D, N]
    float4* ho = (float4*)(h_out + (long)b * D_MODEL * D_STATE + d * D_STATE);
    #pragma unroll
    for (int q = 0; q < 16; ++q) ho[q] = hrow[q];
}

// ---------------------------------------------------------------------------
extern "C" void kernel_launch(void* const* d_in, const int* in_sizes, int n_in,
                              void* d_out, int out_size, void* d_ws, size_t ws_size,
                              hipStream_t stream) {
    const float* x     = (const float*)d_in[0];   // [8,4096,1024]
    const float* A_raw = (const float*)d_in[1];   // [1024,64]
    const float* B_w   = (const float*)d_in[2];   // [64,1024]
    const float* C_w   = (const float*)d_in[3];   // [1024,64]
    const float* gamma = (const float*)d_in[4];   // [1024]

    float* y_out = (float*)d_out;                               // [8,4096,1024]
    float* h_out = y_out + (size_t)B_BATCH * SEQ * D_MODEL;     // [8,1024,64]

    // workspace layout (floats): A_s | C_s | g_s | Bm_h(f16) | U
    float*    A_s  = (float*)d_ws;                       // 65536 f32
    float*    C_s  = A_s + D_MODEL * D_STATE;            // 65536 f32
    float*    g_s  = C_s + D_MODEL * D_STATE;            // 1024  f32
    _Float16* Bm_h = (_Float16*)(g_s + D_MODEL);         // 65536 f16 (32B-aligned)
    float*    U    = (float*)(Bm_h + D_STATE * D_MODEL); // 32768*64 f32

    prep_kernel<<<(D_MODEL * D_STATE + 255) / 256, 256, 0, stream>>>(
        A_raw, B_w, C_w, gamma, A_s, Bm_h, C_s, g_s);

    gemm_u_kernel<<<(B_BATCH * SEQ) / 128, 256, 0, stream>>>(x, Bm_h, U);

    const size_t lds_bytes = (size_t)RED_OFF + 64 * sizeof(float);
    scan_kernel<<<B_BATCH, D_MODEL, lds_bytes, stream>>>(
        x, A_s, C_s, g_s, U, y_out, h_out);
}
